// MSDeformAttn_30288109372172
// MI455X (gfx1250) — compile-verified, matrix operand order checked by reading, not run on hardware
//
#include <hip/hip_runtime.h>
#include <hip/hip_bf16.h>
#include <math.h>

// ---------------------------------------------------------------------------
// MSDeformAttn for MI455X (gfx1250, wave32, WMMA).
//
// GEMMs run on v_wmma_f32_16x16x32_f16 with an f16 hi/lo split (x = h + l,
// A*B ~= AhBh + AhBl + AlBh, ~2^-22 relative error == f32-quality).
// All f32->f16 splitting is hoisted into pre-pack kernels that emit fragments
// in the exact WMMA VGPR layouts, so the GEMM inner loop is just
// global_load_b128 (A) + ds_load_b128 (B from LDS) + v_wmma.
// Sampling is a wave-per-(b,q,head) gather with lane = channel (Dh=32) so
// each corner fetch is one contiguous 128B load.
// ---------------------------------------------------------------------------

typedef __attribute__((ext_vector_type(16))) _Float16 v16h;
typedef __attribute__((ext_vector_type(8)))  float    v8f;
typedef __attribute__((ext_vector_type(4)))  float    v4f;

#define LQ      13294
#define LIN     13294
#define NB      2               // N * NF
#define DMODEL  256
#define KDIM    256
#define NHEAD   8
#define DH      32
#define NLVL    4
#define NPTS    4
#define M_TOTAL (NB * LQ)       // 26588 rows in every GEMM
#define MTILES     ((M_TOTAL + 15) / 16)   // 1662
#define MBLOCKS    ((MTILES + 7) / 8)      // 208
#define MTILES_PAD (MBLOCKS * 8)           // 1664
#define KCHUNKS    (KDIM / 32)             // 8
// fragment = 32 lanes x 16 halfs = 512 halfs; hi/lo planes -> 1024 per chunk
#define FRAG       512

__device__ __forceinline__ void split16(const float* __restrict__ x, v16h& hi, v16h& lo) {
#pragma unroll
    for (int j = 0; j < 16; ++j) {
        float    v = x[j];
        _Float16 h = (_Float16)v;
        hi[j] = h;
        lo[j] = (_Float16)(v - (float)h);
    }
}

// ---- Pack W[256,N] into B-fragment layout -----------------------------------
// Wpk[((nt*KCHUNKS + kc)*2 + part)*FRAG + lane*16 + j]
// lane: col = nt*16 + (lane&15); half hf = lane>>4 picks K in
//   {kc*32+8hf .. +7} (j<8) and {kc*32+16+8hf .. +7} (j>=8)   (ISA 7.12.2)
__global__ __launch_bounds__(256)
void pack_w_kernel(const float* __restrict__ W, _Float16* __restrict__ Wpk, int N) {
    const int tid   = blockIdx.x * blockDim.x + threadIdx.x;
    const int total = (N / 16) * KCHUNKS * 32;
    if (tid >= total) return;
    const int lane = tid & 31;
    const int kc   = (tid >> 5) & (KCHUNKS - 1);
    const int nt   = tid >> 8;
    const int hf   = lane >> 4;
    const int l16  = lane & 15;
    const int col  = nt * 16 + l16;

    const float* wp = W + (long)(kc * 32 + 8 * hf) * N + col;
    float vals[16];
#pragma unroll
    for (int j = 0; j < 8; ++j) {
        vals[j]     = wp[(long)j * N];
        vals[j + 8] = wp[(long)(j + 16) * N];
    }
    v16h hi, lo;
    split16(vals, hi, lo);
    _Float16* dst = Wpk + ((long)(nt * KCHUNKS + kc) * 2) * FRAG + lane * 16;
    *(v16h*)dst          = hi;
    *(v16h*)(dst + FRAG) = lo;
}

// ---- Pack A[M,256] into A-fragment layout -----------------------------------
// Apk[((mt*KCHUNKS + kc)*2 + part)*FRAG + lane*16 + j]
// lane: row = mt*16 + (lane&15); same K mapping as B (ISA 7.12.2, 16-bit A).
__global__ __launch_bounds__(256)
void pack_a_kernel(const float* __restrict__ A, _Float16* __restrict__ Apk, int M) {
    const int tid = blockIdx.x * blockDim.x + threadIdx.x;   // one per (mt,kc,lane)
    const int mt  = tid >> 8;
    if (mt >= MTILES_PAD) return;
    const int lane = tid & 31;
    const int kc   = (tid >> 5) & (KCHUNKS - 1);
    const int hf   = lane >> 4;
    const int l16  = lane & 15;

    long row = (long)mt * 16 + l16;
    if (row > M - 1) row = M - 1;                  // clamp; GEMM stores guarded
    const float* ap  = A + row * (long)KDIM + kc * 32 + 8 * hf;
    const v4f*   ap4 = (const v4f*)ap;             // 32B aligned
    float vals[16];
    {
        v4f a0 = ap4[0], a1 = ap4[1], a2 = ap4[4], a3 = ap4[5];
#pragma unroll
        for (int j = 0; j < 4; ++j) {
            vals[j]      = a0[j];
            vals[j + 4]  = a1[j];
            vals[j + 8]  = a2[j];
            vals[j + 12] = a3[j];
        }
    }
    v16h hi, lo;
    split16(vals, hi, lo);
    _Float16* dst = Apk + ((long)(mt * KCHUNKS + kc) * 2) * FRAG + lane * 16;
    *(v16h*)dst          = hi;
    *(v16h*)(dst + FRAG) = lo;
}

// ---- GEMM on packed fragments ----------------------------------------------
// C[M,N] = A @ W + bias. One wave = 16(M) x 64(N); block = 8 waves (128 rows),
// all sharing one 64KB packed-B tile staged in LDS.
#define LDSB_HALFS (4 * KCHUNKS * 2 * FRAG)   // 32768 halfs = 64KB
__global__ __launch_bounds__(256)
void gemm_wmma_packed(const _Float16* __restrict__ Apk, const _Float16* __restrict__ Wpk,
                      const float* __restrict__ bias, float* __restrict__ C,
                      int M, int N) {
    __shared__ __align__(16) _Float16 ldsB[LDSB_HALFS];

    // cooperative copy of this block's 4 N-tiles of packed B into LDS
    const int nt0 = blockIdx.y * 4;
    {
        const uint4* src = (const uint4*)(Wpk + (long)nt0 * KCHUNKS * 2 * FRAG);
        uint4*       dst = (uint4*)ldsB;
        for (int i = threadIdx.x; i < LDSB_HALFS / 8; i += 256) dst[i] = src[i];
    }
    __syncthreads();

    const int lane  = threadIdx.x & 31;
    const int wave  = threadIdx.x >> 5;
    const int mtile = blockIdx.x * 8 + wave;
    const int row0  = mtile * 16;
    const int hf    = lane >> 4;
    const int l16   = lane & 15;
    const int col0  = blockIdx.y * 64;

    const _Float16* arow = Apk + (long)mtile * KCHUNKS * 2 * FRAG + lane * 16;

    v8f acc[4] = {};
#pragma unroll
    for (int kc = 0; kc < KCHUNKS; ++kc) {
        const v16h ah = *(const v16h*)(arow + (kc * 2 + 0) * FRAG);
        const v16h al = *(const v16h*)(arow + (kc * 2 + 1) * FRAG);
        if (kc + 1 < KCHUNKS)
            __builtin_prefetch(arow + (kc * 2 + 2) * FRAG, 0, 0);  // global_prefetch
#pragma unroll
        for (int t = 0; t < 4; ++t) {
            const _Float16* bp = ldsB + ((t * KCHUNKS + kc) * 2) * FRAG + lane * 16;
            const v16h bh = *(const v16h*)bp;
            const v16h bl = *(const v16h*)(bp + FRAG);
            acc[t] = __builtin_amdgcn_wmma_f32_16x16x32_f16(
                false, ah, false, bh, (short)0, acc[t], false, false);
            acc[t] = __builtin_amdgcn_wmma_f32_16x16x32_f16(
                false, ah, false, bl, (short)0, acc[t], false, false);
            acc[t] = __builtin_amdgcn_wmma_f32_16x16x32_f16(
                false, al, false, bh, (short)0, acc[t], false, false);
        }
    }

    // C/D layout: col = lane&15, row = vgpr + 8*(lane>>4)   (ISA 7.12.2)
#pragma unroll
    for (int t = 0; t < 4; ++t) {
        const int col = col0 + 16 * t + l16;
        const float b = bias[col];
#pragma unroll
        for (int v = 0; v < 8; ++v) {
            const int row = row0 + v + 8 * hf;
            if (row < M) C[(long)row * N + col] = acc[t][v] + b;
        }
    }
}

// ---- softmax + sampling locations (in place) --------------------------------
__global__ __launch_bounds__(256)
void prep_loc_attn(float* __restrict__ offbuf, float* __restrict__ attnbuf,
                   const float* __restrict__ refpts, const int* __restrict__ spatial) {
    const int tid   = blockIdx.x * blockDim.x + threadIdx.x;
    const int total = M_TOTAL * NHEAD;
    if (tid >= total) return;
    const int h  = tid & 7;
    const int bq = tid >> 3;
    const int q  = bq % LQ;

    float* ab = attnbuf + (long)bq * (NHEAD * NLVL * NPTS) + h * 16;
    float  vals[16];
    float  m = -1e30f;
#pragma unroll
    for (int i = 0; i < 16; ++i) { vals[i] = ab[i]; m = fmaxf(m, vals[i]); }
    float s = 0.f;
#pragma unroll
    for (int i = 0; i < 16; ++i) { vals[i] = expf(vals[i] - m); s += vals[i]; }
    const float inv = 1.f / s;
#pragma unroll
    for (int i = 0; i < 16; ++i) ab[i] = vals[i] * inv;

    float* ob = offbuf + (long)bq * DMODEL + h * (NLVL * NPTS * 2);
#pragma unroll
    for (int l = 0; l < NLVL; ++l) {
        const float Hs = (float)spatial[2 * l + 0];
        const float Ws = (float)spatial[2 * l + 1];
        const float rx = refpts[((long)q * NLVL + l) * 2 + 0];
        const float ry = refpts[((long)q * NLVL + l) * 2 + 1];
#pragma unroll
        for (int p = 0; p < NPTS; ++p) {
            const int o = (l * NPTS + p) * 2;
            ob[o + 0] = rx + ob[o + 0] / Ws;
            ob[o + 1] = ry + ob[o + 1] / Hs;
        }
    }
}

__device__ __forceinline__ float corner_val(const float* __restrict__ vbase, int start,
                                            int Ws, int Hs, int xi, int yi, float w) {
    if (xi < 0 || xi >= Ws || yi < 0 || yi >= Hs) return 0.f;
    const long idx = start + (long)yi * Ws + xi;
    return w * vbase[idx * DMODEL];
}

// ---- deformable bilinear gather: one wave per (b,q,h), lane = channel -------
__global__ __launch_bounds__(256)
void msda_sample(const float* __restrict__ vbuf, const float* __restrict__ locbuf,
                 const float* __restrict__ awbuf, const int* __restrict__ spatial,
                 const int* __restrict__ lsi, float* __restrict__ out) {
    const int lane = threadIdx.x & 31;
    const int wid  = (blockIdx.x * blockDim.x + threadIdx.x) >> 5;
    if (wid >= M_TOTAL * NHEAD) return;
    const int h  = wid & 7;
    const int bq = wid >> 3;
    const int b  = bq / LQ;

    const float* loc   = locbuf + (long)bq * DMODEL + h * (NLVL * NPTS * 2);
    const float* aw    = awbuf + (long)bq * (NHEAD * NLVL * NPTS) + h * 16;
    const float* vbase = vbuf + (long)b * LIN * DMODEL + h * DH + lane;

    float acc = 0.f;
#pragma unroll
    for (int l = 0; l < NLVL; ++l) {
        const int Hs    = spatial[2 * l + 0];
        const int Ws    = spatial[2 * l + 1];
        const int start = lsi[l];
#pragma unroll
        for (int p = 0; p < NPTS; ++p) {
            const float x  = loc[(l * NPTS + p) * 2 + 0] * (float)Ws - 0.5f;
            const float y  = loc[(l * NPTS + p) * 2 + 1] * (float)Hs - 0.5f;
            const float fx = floorf(x), fy = floorf(y);
            const float lx = x - fx, ly = y - fy;
            const int   x0 = (int)fx, y0 = (int)fy;
            const float w  = aw[l * NPTS + p];
            acc += w * corner_val(vbase, start, Ws, Hs, x0,     y0,     (1.f - lx) * (1.f - ly));
            acc += w * corner_val(vbase, start, Ws, Hs, x0 + 1, y0,     lx * (1.f - ly));
            acc += w * corner_val(vbase, start, Ws, Hs, x0,     y0 + 1, (1.f - lx) * ly);
            acc += w * corner_val(vbase, start, Ws, Hs, x0 + 1, y0 + 1, lx * ly);
        }
    }
    out[(long)bq * DMODEL + h * DH + lane] = acc;
}

extern "C" void kernel_launch(void* const* d_in, const int* in_sizes, int n_in,
                              void* d_out, int out_size, void* d_ws, size_t ws_size,
                              hipStream_t stream) {
    const float* query   = (const float*)d_in[0];
    const float* refpts  = (const float*)d_in[1];
    const float* flat    = (const float*)d_in[2];
    const int*   spatial = (const int*)d_in[3];
    const int*   lsi     = (const int*)d_in[4];
    const float* Wv      = (const float*)d_in[5];
    const float* bv      = (const float*)d_in[6];
    const float* Woff    = (const float*)d_in[7];
    const float* boff    = (const float*)d_in[8];
    const float* Wattn   = (const float*)d_in[9];
    const float* battn   = (const float*)d_in[10];
    const float* Wout    = (const float*)d_in[11];
    const float* bout    = (const float*)d_in[12];
    float*       outp    = (float*)d_out;

    // ws layout (floats): value | off->loc | attn->aw | sampled | packs
    float* ws      = (float*)d_ws;
    float* vbuf    = ws;                                   // 26588*256
    float* offbuf  = vbuf + (long)M_TOTAL * DMODEL;        // 26588*256
    float* attnbuf = offbuf + (long)M_TOTAL * DMODEL;      // 26588*128
    float* sampled = attnbuf + (long)M_TOTAL * (NHEAD * NLVL * NPTS);
    _Float16* apk      = (_Float16*)(sampled + (long)M_TOTAL * DMODEL);
    _Float16* wpk_v    = apk + (long)MTILES_PAD * KCHUNKS * 2 * FRAG;   // 13.6M halfs
    _Float16* wpk_off  = wpk_v + (DMODEL / 16) * KCHUNKS * 2 * FRAG;    // 128K halfs
    _Float16* wpk_attn = wpk_off + (DMODEL / 16) * KCHUNKS * 2 * FRAG;
    _Float16* wpk_out  = wpk_attn + (128 / 16) * KCHUNKS * 2 * FRAG;

    const dim3 blk(256);
    const int  wthreads256 = (DMODEL / 16) * KCHUNKS * 32;   // 4096
    const int  wthreads128 = (128 / 16) * KCHUNKS * 32;      // 2048
    const int  ablocks     = MTILES_PAD;                     // (MTILES_PAD*256)/256

    // pack all weight matrices (tiny)
    pack_w_kernel<<<(wthreads256 + 255) / 256, blk, 0, stream>>>(Wv, wpk_v, DMODEL);
    pack_w_kernel<<<(wthreads256 + 255) / 256, blk, 0, stream>>>(Woff, wpk_off, DMODEL);
    pack_w_kernel<<<(wthreads128 + 255) / 256, blk, 0, stream>>>(Wattn, wpk_attn, 128);
    pack_w_kernel<<<(wthreads256 + 255) / 256, blk, 0, stream>>>(Wout, wpk_out, DMODEL);

    // value = flat @ Wv + bv
    pack_a_kernel<<<ablocks, blk, 0, stream>>>(flat, apk, M_TOTAL);
    gemm_wmma_packed<<<dim3(MBLOCKS, DMODEL / 64), blk, 0, stream>>>(
        apk, wpk_v, bv, vbuf, M_TOTAL, DMODEL);

    // off / attn projections share the packed query
    pack_a_kernel<<<ablocks, blk, 0, stream>>>(query, apk, M_TOTAL);
    gemm_wmma_packed<<<dim3(MBLOCKS, DMODEL / 64), blk, 0, stream>>>(
        apk, wpk_off, boff, offbuf, M_TOTAL, DMODEL);
    gemm_wmma_packed<<<dim3(MBLOCKS, 128 / 64), blk, 0, stream>>>(
        apk, wpk_attn, battn, attnbuf, M_TOTAL, 128);

    // softmax + loc (in place)
    prep_loc_attn<<<(M_TOTAL * NHEAD + 255) / 256, blk, 0, stream>>>(
        offbuf, attnbuf, refpts, spatial);

    // deformable bilinear gather
    msda_sample<<<(M_TOTAL * NHEAD + 7) / 8, blk, 0, stream>>>(
        vbuf, offbuf, attnbuf, spatial, lsi, sampled);

    // out = sampled @ Wout + bout
    pack_a_kernel<<<ablocks, blk, 0, stream>>>(sampled, apk, M_TOTAL);
    gemm_wmma_packed<<<dim3(MBLOCKS, DMODEL / 64), blk, 0, stream>>>(
        apk, wpk_out, bout, outp, M_TOTAL, DMODEL);
}
